// ContrastiveLoss_23570780520482
// MI455X (gfx1250) — matrix operand order repaired and checked by reading, not compiled, not run on hardware
//
#include <hip/hip_runtime.h>

typedef __attribute__((ext_vector_type(16))) __bf16          v16bf;
typedef __attribute__((ext_vector_type(16))) unsigned short  v16us;
typedef __attribute__((ext_vector_type(8)))  unsigned short  v8us;
typedef __attribute__((ext_vector_type(8)))  float           v8f;

#define DIM      128
#define KCHUNKS  4                        // 128 / 32
#define WPB      4                        // waves per block
#define MBLK     2                        // i-tiles per wave (register M-blocking)
#define LDSPITCH 136                      // 128 + 8 u16 pad -> 272B pitch, conflict-free b128
#define TEMP_INV 14.285714285714285714f   // 1 / 0.07

// ---- fp32 -> bf16 RNE, manual bit path
__device__ __forceinline__ unsigned short f2bf(float x) {
  unsigned u = __builtin_bit_cast(unsigned, x);
  unsigned r = (u + 0x7FFFu + ((u >> 16) & 1u)) >> 16;
  return (unsigned short)r;
}
__device__ __forceinline__ float bf2f(unsigned short b) {
  return __builtin_bit_cast(float, ((unsigned)b) << 16);
}

// 8 consecutive floats -> hi/lo bf16 at vector elements [base, base+8)
__device__ __forceinline__ void cvt8(const float* __restrict__ s,
                                     v16us& hi, v16us& lo, int base) {
  float4 u0 = *reinterpret_cast<const float4*>(s);
  float4 u1 = *reinterpret_cast<const float4*>(s + 4);
  float x[8] = {u0.x, u0.y, u0.z, u0.w, u1.x, u1.y, u1.z, u1.w};
#pragma unroll
  for (int e = 0; e < 8; ++e) {
    unsigned short hb = f2bf(x[e]);
    hi[base + e] = hb;
    lo[base + e] = f2bf(x[e] - bf2f(hb));   // x - hi(x) exact in fp32
  }
}

// 32B LDS read as two aligned 16B ops -> one WMMA bf16 operand
__device__ __forceinline__ v16bf lds_read16(const unsigned short* p) {
  v8us a = *reinterpret_cast<const v8us*>(p);
  v8us b = *reinterpret_cast<const v8us*>(p + 8);
  v16us f = __builtin_shufflevector(a, b, 0,1,2,3,4,5,6,7,8,9,10,11,12,13,14,15);
  return __builtin_bit_cast(v16bf, f);
}

__global__ void __launch_bounds__(WPB * 32)
contrastive_main(const float* __restrict__ emb, const int* __restrict__ labels,
                 int B, int JS, int nblk_i,
                 float* __restrict__ se_p, float* __restrict__ ps_p,
                 int* __restrict__ pc_p) {
  // [buffer][hi/lo][row][k], double buffered, padded pitch
  __shared__ __align__(16) unsigned short lds[2][2][16][LDSPITCH];

  const int ntiles   = B >> 4;
  const int ib       = blockIdx.x % nblk_i;   // i-block (fastest -> j-range sharing in L2)
  const int split    = blockIdx.x / nblk_i;   // j-range split
  const int tilesPer = ntiles / JS;
  const int jt0 = split * tilesPer;
  const int jt1 = jt0 + tilesPer;

  const int w    = threadIdx.x >> 5;
  const int lane = threadIdx.x & 31;
  const int h    = lane >> 4;                 // lane half
  const int q    = lane & 15;                 // M (A loads) / N (B reads, C cols)

  // staging role: 128 threads cover 16 rows x 8 segments of 16 cols
  const int sr  = threadIdx.x >> 3;
  const int seg = threadIdx.x & 7;

  // ---- per-wave A tiles (MBLK x 16 rows x 128 K), hi/lo bf16 in WMMA A layout
  v16us AhU[MBLK][KCHUNKS], AlU[MBLK][KCHUNKS];
  int   labi[MBLK][8];
  int   ibase[MBLK];
  bool  valid[MBLK];
#pragma unroll
  for (int mb = 0; mb < MBLK; ++mb) {
    int it = ib * (WPB * MBLK) + mb * WPB + w;
    valid[mb] = (it < ntiles);
    if (!valid[mb]) it = ntiles - 1;          // clamp for safe loads
    ibase[mb] = it << 4;
    const float* rowA = emb + (size_t)(ibase[mb] + q) * DIM;
#pragma unroll
    for (int kc = 0; kc < KCHUNKS; ++kc) {
      const int off0 = kc * 32 + h * 16;      // A layout: K = 32kc + 16h + elem
      cvt8(rowA + off0,      AhU[mb][kc], AlU[mb][kc], 0);
      cvt8(rowA + off0 + 8,  AhU[mb][kc], AlU[mb][kc], 8);
    }
#pragma unroll
    for (int v = 0; v < 8; ++v) labi[mb][v] = labels[ibase[mb] + v + 8 * h];
  }

  float se[MBLK][8], ps[MBLK][8];
  int   pc[MBLK][8];
#pragma unroll
  for (int mb = 0; mb < MBLK; ++mb)
#pragma unroll
    for (int v = 0; v < 8; ++v) { se[mb][v] = 0.f; ps[mb][v] = 0.f; pc[mb][v] = 0; }

  // staging helpers
  auto load_tile = [&](int jt_, float4* f) {
    const float4* p =
        reinterpret_cast<const float4*>(emb + (size_t)((jt_ << 4) + sr) * DIM + seg * 16);
    f[0] = p[0]; f[1] = p[1]; f[2] = p[2]; f[3] = p[3];
  };
  auto cvt_store = [&](int buf, const float4* f) {
    float x[16] = {f[0].x, f[0].y, f[0].z, f[0].w, f[1].x, f[1].y, f[1].z, f[1].w,
                   f[2].x, f[2].y, f[2].z, f[2].w, f[3].x, f[3].y, f[3].z, f[3].w};
    v16us hv, lv;
#pragma unroll
    for (int e = 0; e < 16; ++e) {
      unsigned short hb = f2bf(x[e]);
      hv[e] = hb;
      lv[e] = f2bf(x[e] - bf2f(hb));
    }
    unsigned short* dh = &lds[buf][0][sr][seg * 16];
    unsigned short* dl = &lds[buf][1][sr][seg * 16];
    *reinterpret_cast<v8us*>(dh)     = __builtin_shufflevector(hv, hv, 0,1,2,3,4,5,6,7);
    *reinterpret_cast<v8us*>(dh + 8) = __builtin_shufflevector(hv, hv, 8,9,10,11,12,13,14,15);
    *reinterpret_cast<v8us*>(dl)     = __builtin_shufflevector(lv, lv, 0,1,2,3,4,5,6,7);
    *reinterpret_cast<v8us*>(dl + 8) = __builtin_shufflevector(lv, lv, 8,9,10,11,12,13,14,15);
  };

  // prologue: stage first tile
  {
    float4 f[4];
    load_tile(jt0, f);
    cvt_store(0, f);
  }
  __syncthreads();

  for (int jt = jt0; jt < jt1; ++jt) {
    const int cur = (jt - jt0) & 1;
    const bool have_next = (jt + 1 < jt1);
    float4 nf[4];
    if (have_next) load_tile(jt + 1, nf);     // issue early; consumed after compute

    const int j_base = jt << 4;
    const int labj = labels[j_base + q];

    v8f C[MBLK];
#pragma unroll
    for (int mb = 0; mb < MBLK; ++mb) C[mb] = (v8f){0.f,0.f,0.f,0.f,0.f,0.f,0.f,0.f};

#pragma unroll
    for (int kc = 0; kc < KCHUNKS; ++kc) {
      const int off = kc * 32 + h * 16;       // B layout: lane=N, K = 32kc + 16h + elem
      v16bf Bh = lds_read16(&lds[cur][0][q][off]);
      v16bf Bl = lds_read16(&lds[cur][1][q][off]);
#pragma unroll
      for (int mb = 0; mb < MBLK; ++mb) {
        v16bf Ah = __builtin_bit_cast(v16bf, AhU[mb][kc]);
        v16bf Al = __builtin_bit_cast(v16bf, AlU[mb][kc]);
        C[mb] = __builtin_amdgcn_wmma_f32_16x16x32_bf16(false, Ah, false, Bh, (short)0, C[mb], false, false);
        C[mb] = __builtin_amdgcn_wmma_f32_16x16x32_bf16(false, Ah, false, Bl, (short)0, C[mb], false, false);
        C[mb] = __builtin_amdgcn_wmma_f32_16x16x32_bf16(false, Al, false, Bh, (short)0, C[mb], false, false);
      }
    }

    // C layout: lane = column N = q, rows M = v + 8h
    const int jg = j_base + q;
#pragma unroll
    for (int mb = 0; mb < MBLK; ++mb) {
#pragma unroll
      for (int v = 0; v < 8; ++v) {
        const int ig = ibase[mb] + v + 8 * h;
        const float s = C[mb][v] * TEMP_INV;
        if (valid[mb] && ig != jg) {          // diagonal == -1e9 -> exp 0
          se[mb][v] += __expf(s);             // bounded by e^14.3
          if (labj == labi[mb][v]) { ps[mb][v] += s; pc[mb][v] += 1; }
        }
      }
    }

    if (have_next) cvt_store(cur ^ 1, nf);    // convert + stage behind compute
    __syncthreads();
  }

  // reduce across the 16 column-lanes of each half (xor masks stay in-half)
#pragma unroll
  for (int m = 1; m < 16; m <<= 1) {
#pragma unroll
    for (int mb = 0; mb < MBLK; ++mb)
#pragma unroll
      for (int v = 0; v < 8; ++v) {
        se[mb][v] += __shfl_xor(se[mb][v], m, 32);
        ps[mb][v] += __shfl_xor(ps[mb][v], m, 32);
        pc[mb][v] += __shfl_xor(pc[mb][v], m, 32);
      }
  }

  if (q == 0) {   // lanes 0 / 16 write rows 0..7 / 8..15 of each i-tile
#pragma unroll
    for (int mb = 0; mb < MBLK; ++mb) {
      if (!valid[mb]) continue;
#pragma unroll
      for (int v = 0; v < 8; ++v) {
        const int row = ibase[mb] + v + 8 * h;
        const size_t slot = (size_t)row * JS + split;
        se_p[slot] = se[mb][v];
        ps_p[slot] = ps[mb][v];
        pc_p[slot] = pc[mb][v];
      }
    }
  }
}

// Deterministic fixed-order final reduction
__global__ void __launch_bounds__(256)
contrastive_finalize(const float* __restrict__ se_p, const float* __restrict__ ps_p,
                     const int* __restrict__ pc_p, int B, int JS,
                     float* __restrict__ out) {
  __shared__ float s_sum[256];
  __shared__ float s_cnt[256];
  float acc = 0.f, cnt = 0.f;
  for (int r = threadIdx.x; r < B; r += 256) {
    float se = 0.f, ps = 0.f; int pc = 0;
    for (int s = 0; s < JS; ++s) {
      se += se_p[(size_t)r * JS + s];
      ps += ps_p[(size_t)r * JS + s];
      pc += pc_p[(size_t)r * JS + s];
    }
    if (pc > 0) { acc += ps / (float)pc - __logf(se); cnt += 1.f; }
  }
  s_sum[threadIdx.x] = acc;
  s_cnt[threadIdx.x] = cnt;
  __syncthreads();
  for (int off = 128; off > 0; off >>= 1) {
    if (threadIdx.x < off) {
      s_sum[threadIdx.x] += s_sum[threadIdx.x + off];
      s_cnt[threadIdx.x] += s_cnt[threadIdx.x + off];
    }
    __syncthreads();
  }
  if (threadIdx.x == 0) out[0] = -(s_sum[0] / fmaxf(s_cnt[0], 1.f));
}

extern "C" void kernel_launch(void* const* d_in, const int* in_sizes, int n_in,
                              void* d_out, int out_size, void* d_ws, size_t ws_size,
                              hipStream_t stream) {
  const float* emb    = (const float*)d_in[0];
  const int*   labels = (const int*)d_in[1];
  const int B = in_sizes[1];           // 8192; D fixed at 128 per reference
  const int ntiles = B >> 4;

  // j-range split factor (extra parallelism); partials live in d_ws
  int JS = 8;
  if (JS > ntiles) JS = 1;
  while (JS > 1 && (((size_t)B * JS * 12 > ws_size) || (ntiles % JS) != 0)) JS >>= 1;

  float* se_p = (float*)d_ws;
  float* ps_p = se_p + (size_t)B * JS;
  int*   pc_p = (int*)(ps_p + (size_t)B * JS);

  const int nblk_i = (ntiles + WPB * MBLK - 1) / (WPB * MBLK);
  dim3 block(WPB * 32);
  dim3 grid(nblk_i * JS);

  contrastive_main<<<grid, block, 0, stream>>>(emb, labels, B, JS, nblk_i,
                                               se_p, ps_p, pc_p);
  contrastive_finalize<<<1, 256, 0, stream>>>(se_p, ps_p, pc_p, B, JS, (float*)d_out);
}
